// GAT_31164282699922
// MI455X (gfx1250) — matrix operand order repaired and checked by reading, not compiled, not run on hardware
//
#include <hip/hip_runtime.h>
#include <stdint.h>

// GATv2 (3 layers, 4 heads, hid 64) for gfx1250.
// - GEMMs (h@[Wl|Wr]) run on v_wmma_f32_16x16x32_bf16, f32 accumulate.
// - Edge phase: wave-per-edge coalesced gather + wave32 shuffle reduction,
//   segment softmax via ordered-uint atomicMax + f32 atomicAdd (native on CDNA5 L2),
//   block-per-edge scatter with consecutive-address f32 atomics.

#define NSLOPE 0.2f

typedef __attribute__((ext_vector_type(16))) __bf16 v16bf;
typedef __attribute__((ext_vector_type(8)))  float  v8f;

union FragAB { v16bf v; uint4 q[2]; };
union FragC  { v8f   v; float f[8]; };

__device__ __forceinline__ unsigned short f32_to_bf16(float f) {
    uint32_t u = __float_as_uint(f);
    uint32_t r = u + 0x7FFFu + ((u >> 16) & 1u);   // round-to-nearest-even
    return (unsigned short)(r >> 16);
}
// Order-preserving float<->uint key for atomicMax on floats (incl. negatives)
__device__ __forceinline__ unsigned int fkey(float f) {
    unsigned int u = __float_as_uint(f);
    return (u & 0x80000000u) ? ~u : (u | 0x80000000u);
}
__device__ __forceinline__ float fdec(unsigned int k) {
    unsigned int u = (k & 0x80000000u) ? (k & 0x7FFFFFFFu) : ~k;
    return __uint_as_float(u);
}
__device__ __forceinline__ float lrelu(float x) { return x > 0.f ? x : x * NSLOPE; }

// ---------------- conversions ----------------
__global__ void cvt_bf16(const float* __restrict__ in, unsigned short* __restrict__ out, int n) {
    int i = blockIdx.x * blockDim.x + threadIdx.x;
    if (i < n) out[i] = f32_to_bf16(in[i]);
}

// Wt[c*K + k] = bf16( c<256 ? Wl[k][c] : Wr[k][c-256] )  (transpose for contiguous-K B frags)
__global__ void cvt_weights(const float* __restrict__ Wl, const float* __restrict__ Wr,
                            unsigned short* __restrict__ Wt, int K) {
    int idx = blockIdx.x * blockDim.x + threadIdx.x;
    if (idx >= 512 * K) return;
    int c = idx / K, k = idx - c * K;
    float v = (c < 256) ? Wl[(size_t)k * 256 + c] : Wr[(size_t)k * 256 + (c - 256)];
    Wt[idx] = f32_to_bf16(v);
}

// ---------------- WMMA GEMM: C[M x 512] = A[M x K] * B[K x 512] ----------------
// A row-major bf16 (lda = K); B given as Bt[512 x K] row-major bf16; C f32 row-major (ldc=512).
// 8 waves/block, each wave owns one 16x16 output tile.
template <int K>
__global__ void gemm_bf16_wmma(const unsigned short* __restrict__ A,
                               const unsigned short* __restrict__ Bt,
                               float* __restrict__ C, int M) {
    const int lane  = threadIdx.x & 31;
    const int wave  = threadIdx.x >> 5;
    const int ntile = blockIdx.x;                 // 0..31   (512 cols)
    const int mtile = blockIdx.y * 8 + wave;
    if (mtile * 16 >= M) return;

    int row = mtile * 16 + (lane & 15);           // A row for this lane's frag
    if (row >= M) row = M - 1;                    // defensive clamp (M=50000 is /16)
    const int col = ntile * 16 + (lane & 15);     // B column for this lane's frag
    const int kA  = (lane >> 4) * 8;              // A: lanes 0-15 -> K 0..7/16..23, 16-31 -> 8..15/24..31
    const int kB  = (lane >> 4) * 8;              // B: lanes 0-15 -> K 0..7/16..23, 16-31 -> 8..15/24..31

    const uint4* Arow = reinterpret_cast<const uint4*>(A + (size_t)row * K);
    const uint4* Brow = reinterpret_cast<const uint4*>(Bt + (size_t)col * K);

    FragC acc;
    #pragma unroll
    for (int r = 0; r < 8; ++r) acc.f[r] = 0.f;

    #pragma unroll
    for (int k0 = 0; k0 < K; k0 += 32) {
        FragAB a, b;
        a.q[0] = Arow[(k0 + kA) >> 3];
        a.q[1] = Arow[(k0 + kA + 16) >> 3];
        b.q[0] = Brow[(k0 + kB) >> 3];
        b.q[1] = Brow[(k0 + kB + 16) >> 3];
        acc.v = __builtin_amdgcn_wmma_f32_16x16x32_bf16(
            false, a.v, false, b.v, (short)0, acc.v, false, false);
    }

    // D layout: vgpr r -> row (lane<16 ? r : r+8), lane&15 -> col
    const int r0 = (lane >> 4) * 8;
    float* Cout = C + (size_t)(mtile * 16 + r0) * 512 + (size_t)ntile * 16 + (lane & 15);
    #pragma unroll
    for (int r = 0; r < 8; ++r) Cout[(size_t)r * 512] = acc.f[r];
}

// ---------------- edge phase ----------------
// e[e][h] = sum_c lrelu(xl[s][h][c] + xr[d][h][c]) * att[h][c]; atomicMax per dst
__global__ void edge_logits(const int* __restrict__ src, const int* __restrict__ dst,
                            const float* __restrict__ xlxr, const float* __restrict__ att,
                            float* __restrict__ ebuf, unsigned int* __restrict__ emax, int E) {
    const int wave = threadIdx.x >> 5, lane = threadIdx.x & 31;
    const int e = blockIdx.x * 8 + wave;
    if (e >= E) return;
    const int s = src[e], d = dst[e];
    const float* xs = xlxr + (size_t)s * 512;         // xl part
    const float* xd = xlxr + (size_t)d * 512 + 256;   // xr part
    float a[4];
    #pragma unroll
    for (int h = 0; h < 4; ++h) {
        float acc = 0.f;
        #pragma unroll
        for (int cc = 0; cc < 2; ++cc) {
            int c = lane + cc * 32;
            acc += lrelu(xs[h * 64 + c] + xd[h * 64 + c]) * att[h * 64 + c];
        }
        a[h] = acc;
    }
    #pragma unroll
    for (int off = 16; off > 0; off >>= 1) {
        #pragma unroll
        for (int h = 0; h < 4; ++h) a[h] += __shfl_down(a[h], off, 32);
    }
    if (lane == 0) {
        #pragma unroll
        for (int h = 0; h < 4; ++h) {
            ebuf[(size_t)e * 4 + h] = a[h];
            atomicMax(&emax[(size_t)d * 4 + h], fkey(a[h]));
        }
    }
}

__global__ void self_logits(const float* __restrict__ xlxr, const float* __restrict__ att,
                            float* __restrict__ eself, unsigned int* __restrict__ emax, int N) {
    const int wave = threadIdx.x >> 5, lane = threadIdx.x & 31;
    const int i = blockIdx.x * 8 + wave;
    if (i >= N) return;
    const float* xs = xlxr + (size_t)i * 512;
    const float* xd = xs + 256;
    float a[4];
    #pragma unroll
    for (int h = 0; h < 4; ++h) {
        float acc = 0.f;
        #pragma unroll
        for (int cc = 0; cc < 2; ++cc) {
            int c = lane + cc * 32;
            acc += lrelu(xs[h * 64 + c] + xd[h * 64 + c]) * att[h * 64 + c];
        }
        a[h] = acc;
    }
    #pragma unroll
    for (int off = 16; off > 0; off >>= 1) {
        #pragma unroll
        for (int h = 0; h < 4; ++h) a[h] += __shfl_down(a[h], off, 32);
    }
    if (lane == 0) {
        #pragma unroll
        for (int h = 0; h < 4; ++h) {
            eself[(size_t)i * 4 + h] = a[h];
            atomicMax(&emax[(size_t)i * 4 + h], fkey(a[h]));
        }
    }
}

__global__ void edge_softmax_num(const int* __restrict__ dst, float* __restrict__ ebuf,
                                 const unsigned int* __restrict__ emax,
                                 float* __restrict__ denom, int total) {
    int idx = blockIdx.x * blockDim.x + threadIdx.x;
    if (idx >= total) return;
    int e = idx >> 2, h = idx & 3;
    int d = dst[e];
    float ee = __expf(ebuf[idx] - fdec(emax[(size_t)d * 4 + h]));
    ebuf[idx] = ee;
    atomicAdd(&denom[(size_t)d * 4 + h], ee);
}

__global__ void self_softmax_num(float* __restrict__ eself, const unsigned int* __restrict__ emax,
                                 float* __restrict__ denom, int total) {
    int idx = blockIdx.x * blockDim.x + threadIdx.x;
    if (idx >= total) return;
    float ee = __expf(eself[idx] - fdec(emax[idx]));
    eself[idx] = ee;
    atomicAdd(&denom[idx], ee);
}

// out[d][h][c] += xl[s][h][c] * alpha ; one block per edge, 256 consecutive channels
__global__ void edge_aggregate(const int* __restrict__ src, const int* __restrict__ dst,
                               const float* __restrict__ xlxr, const float* __restrict__ ebuf,
                               const float* __restrict__ denom, float* __restrict__ out) {
    const int e = blockIdx.x;
    const int j = threadIdx.x;          // 0..255, h = j>>6, c = j&63
    const int h = j >> 6;
    const int s = src[e], d = dst[e];
    const float alpha = ebuf[(size_t)e * 4 + h] / denom[(size_t)d * 4 + h];
    atomicAdd(&out[(size_t)d * 256 + j], xlxr[(size_t)s * 512 + j] * alpha);
}

__global__ void self_aggregate(const float* __restrict__ xlxr, const float* __restrict__ eself,
                               const float* __restrict__ denom, float* __restrict__ out) {
    const int i = blockIdx.x;
    const int j = threadIdx.x;
    const int h = j >> 6;
    const float alpha = eself[(size_t)i * 4 + h] / denom[(size_t)i * 4 + h];
    atomicAdd(&out[(size_t)i * 256 + j], xlxr[(size_t)i * 512 + j] * alpha);
}

// mean over heads + bias -> layernorm -> leaky_relu ; one wave per node, 2 channels/lane
__global__ void finalize(const float* __restrict__ out, const float* __restrict__ bias,
                         const float* __restrict__ g, const float* __restrict__ b,
                         float* __restrict__ hout, int N) {
    const int wave = threadIdx.x >> 5, lane = threadIdx.x & 31;
    const int i = blockIdx.x * 8 + wave;
    if (i >= N) return;
    float v0 = 0.f, v1 = 0.f;
    #pragma unroll
    for (int h = 0; h < 4; ++h) {
        v0 += out[(size_t)i * 256 + h * 64 + lane];
        v1 += out[(size_t)i * 256 + h * 64 + lane + 32];
    }
    v0 = v0 * 0.25f + bias[lane];
    v1 = v1 * 0.25f + bias[lane + 32];
    float s1 = v0 + v1, s2 = v0 * v0 + v1 * v1;
    #pragma unroll
    for (int off = 16; off > 0; off >>= 1) {
        s1 += __shfl_xor(s1, off, 32);
        s2 += __shfl_xor(s2, off, 32);
    }
    const float mu  = s1 * (1.f / 64.f);
    const float var = s2 * (1.f / 64.f) - mu * mu;
    const float inv = rsqrtf(var + 1e-5f);
    float y0 = (v0 - mu) * inv * g[lane] + b[lane];
    float y1 = (v1 - mu) * inv * g[lane + 32] + b[lane + 32];
    hout[(size_t)i * 64 + lane]      = lrelu(y0);
    hout[(size_t)i * 64 + lane + 32] = lrelu(y1);
}

// ---------------- driver ----------------
extern "C" void kernel_launch(void* const* d_in, const int* in_sizes, int n_in,
                              void* d_out, int out_size, void* d_ws, size_t ws_size,
                              hipStream_t stream) {
    const float* x  = (const float*)d_in[0];
    const int*   ei = (const int*)d_in[1];
    const int N = in_sizes[0] / 128;
    const int E = in_sizes[1] / 2;
    const int* src = ei;
    const int* dst = ei + E;
    const int Mpad = ((N + 15) / 16) * 16;

    char* w = (char*)d_ws;
    size_t off = 0;
    auto alloc = [&](size_t bytes) -> char* {
        char* p = w + off;
        off += (bytes + 255) & ~(size_t)255;
        return p;
    };
    float*          xlxr  = (float*)alloc((size_t)Mpad * 512 * 4);
    unsigned short* hbf   = (unsigned short*)alloc((size_t)N * 128 * 2);
    unsigned short* Wt    = (unsigned short*)alloc((size_t)512 * 128 * 2);
    float*          ebuf  = (float*)alloc((size_t)E * 4 * 4);
    float*          eself = (float*)alloc((size_t)N * 4 * 4);
    unsigned int*   emax  = (unsigned int*)alloc((size_t)N * 4 * 4);
    float*          denom = (float*)alloc((size_t)N * 4 * 4);
    float*          outac = (float*)alloc((size_t)N * 256 * 4);
    float*          ha    = (float*)alloc((size_t)N * 64 * 4);
    float*          hb    = (float*)alloc((size_t)N * 64 * 4);

    const float* hcur = x;
    for (int layer = 0; layer < 3; ++layer) {
        const int K = (layer == 0) ? 128 : 64;
        const float* Wl   = (const float*)d_in[2 + layer * 6 + 0];
        const float* Wr   = (const float*)d_in[2 + layer * 6 + 1];
        const float* att  = (const float*)d_in[2 + layer * 6 + 2];
        const float* bias = (const float*)d_in[2 + layer * 6 + 3];
        const float* lng  = (const float*)d_in[2 + layer * 6 + 4];
        const float* lnb  = (const float*)d_in[2 + layer * 6 + 5];
        float* hout = (layer == 2) ? (float*)d_out : ((layer == 0) ? ha : hb);

        cvt_weights<<<(512 * K + 255) / 256, 256, 0, stream>>>(Wl, Wr, Wt, K);
        cvt_bf16<<<(N * K + 255) / 256, 256, 0, stream>>>(hcur, hbf, N * K);

        dim3 gg(32, (unsigned)(((N + 15) / 16 + 7) / 8));
        if (K == 128)
            gemm_bf16_wmma<128><<<gg, 256, 0, stream>>>(hbf, Wt, xlxr, N);
        else
            gemm_bf16_wmma<64><<<gg, 256, 0, stream>>>(hbf, Wt, xlxr, N);

        hipMemsetAsync(emax, 0, (size_t)N * 4 * 4, stream);    // key 0 < key(-inf)
        hipMemsetAsync(denom, 0, (size_t)N * 4 * 4, stream);
        hipMemsetAsync(outac, 0, (size_t)N * 256 * 4, stream);

        edge_logits<<<(E + 7) / 8, 256, 0, stream>>>(src, dst, xlxr, att, ebuf, emax, E);
        self_logits<<<(N + 7) / 8, 256, 0, stream>>>(xlxr, att, eself, emax, N);
        edge_softmax_num<<<(E * 4 + 255) / 256, 256, 0, stream>>>(dst, ebuf, emax, denom, E * 4);
        self_softmax_num<<<(N * 4 + 255) / 256, 256, 0, stream>>>(eself, emax, denom, N * 4);
        edge_aggregate<<<E, 256, 0, stream>>>(src, dst, xlxr, ebuf, denom, outac);
        self_aggregate<<<N, 256, 0, stream>>>(xlxr, eself, denom, outac);
        finalize<<<(N + 7) / 8, 256, 0, stream>>>(outac, bias, lng, lnb, hout, N);

        hcur = hout;
    }
}
